// MetaS5Block_70506183131277
// MI455X (gfx1250) — matrix operand order repaired and checked by reading, not compiled, not run on hardware
//
#include <hip/hip_runtime.h>
#include <hip/hip_bf16.h>

typedef _Float16 h16 __attribute__((ext_vector_type(16)));
typedef _Float16 h8  __attribute__((ext_vector_type(8)));
typedef _Float16 h4  __attribute__((ext_vector_type(4)));
typedef float    v8f __attribute__((ext_vector_type(8)));

#define TOKENS (4 * 4096)
#define LSEQ   4096
#define DIM    1024
#define DIN    512
#define PST    512

#define BM 128
#define BN 128
#define BK 32
#define A_PAD 48    // halfs per LDS row (mult of 8 -> 16B aligned chunks)
#define B_PAD 136   // halfs per LDS row (272B, mult of 16B)

__device__ __forceinline__ float gelu_erf(float x) {
    return 0.5f * x * (1.0f + erff(x * 0.70710678118654752f));
}

// ---------------------------------------------------------------------------
// f16-WMMA GEMM, software pipelined:
//   A tile:  global_load_b128 -> regs (next tile in flight during WMMA) -> cvt -> LDS
//   B tile:  global_load_async_to_lds_b128 (ASYNCcnt), double-buffered LDS
// Cout[M,N] = A(f32,[M,K]) * W(f16,[K,N]) (+ epilogue)
// EPI 0: store.  EPI 1 (S5 out): v = gelu(v + Dv[col]*e1) + e1.  EPI 2: v += e1.
// ---------------------------------------------------------------------------
template <int EPI>
__global__ __launch_bounds__(256) void gemm_f16_kernel(
    const float* __restrict__ A, const _Float16* __restrict__ W,
    float* __restrict__ Cout, int M, int K, int N,
    const float* __restrict__ e1, const float* __restrict__ Dv) {
  __shared__ __align__(16) _Float16 As[2][BM * A_PAD];
  __shared__ __align__(16) _Float16 Bs[2][BK * B_PAD];

  const int tid  = threadIdx.x;
  const int lane = tid & 31;
  const int wid  = tid >> 5;       // 0..7
  const int wm   = wid >> 1;       // 0..3 : 32-row strip
  const int wn   = wid & 1;        // 0..1 : 64-col strip
  const size_t rowBase = (size_t)blockIdx.x * BM;
  const int    colBase = blockIdx.y * BN;

  v8f acc[2][4];
#pragma unroll
  for (int i = 0; i < 2; ++i)
#pragma unroll
    for (int j = 0; j < 4; ++j)
#pragma unroll
      for (int e = 0; e < 8; ++e) acc[i][j][e] = 0.0f;

  // per-thread slots in the staged tiles
  int aRow[4], aCol[4];
#pragma unroll
  for (int i = 0; i < 4; ++i) {
    int lin = tid + i * 256;                  // 0..1023
    aRow[i] = lin >> 3;                       // 0..127
    aCol[i] = (lin & 7) * 4;                  // f32 col
  }
  int bRow[2], bCol[2];
#pragma unroll
  for (int i = 0; i < 2; ++i) {
    int lin = tid + i * 256;                  // 0..511
    bRow[i] = lin >> 4;                       // 0..31 (K row)
    bCol[i] = (lin & 15) * 8;                 // half col
  }

  float4 aReg[4];
  auto loadA = [&](int k0) {
#pragma unroll
    for (int i = 0; i < 4; ++i)
      aReg[i] = *(const float4*)(A + (rowBase + aRow[i]) * (size_t)K + k0 + aCol[i]);
  };
  auto asyncB = [&](int k0, int buf) {
#pragma unroll
    for (int i = 0; i < 2; ++i) {
      const _Float16* src = W + (size_t)(k0 + bRow[i]) * N + colBase + bCol[i];
      unsigned ldsa = (unsigned)(uintptr_t)&Bs[buf][bRow[i] * B_PAD + bCol[i]];
      asm volatile("global_load_async_to_lds_b128 %0, %1, off"
                   :: "v"(ldsa), "v"(src) : "memory");
    }
  };

  loadA(0);
  asyncB(0, 0);

  int buf = 0;
  for (int k0 = 0; k0 < K; k0 += BK) {
    // ---- write phase: staged A regs -> f16 LDS (buf) ----
#pragma unroll
    for (int i = 0; i < 4; ++i) {
      h4 hv = { (_Float16)aReg[i].x, (_Float16)aReg[i].y,
                (_Float16)aReg[i].z, (_Float16)aReg[i].w };
      *(h4*)&As[buf][aRow[i] * A_PAD + aCol[i]] = hv;
    }
    const bool hasNext = (k0 + BK) < K;
    if (hasNext) loadA(k0 + BK);                       // next A tile in flight
    asm volatile("s_wait_asynccnt 0" ::: "memory");    // our Bs[buf] writes done
    __syncthreads();                                   // tile visible; prev reads done
    if (hasNext) asyncB(k0 + BK, buf ^ 1);             // next B tile flies during WMMA

    // ---- fragments per ISA 7.12.2 (16-bit A 16x32; B rows striped by lane) ----
    const int mrow = lane & 15;
    const int hi   = lane >> 4;
    h16 af[2], bfr[4];
#pragma unroll
    for (int mi = 0; mi < 2; ++mi) {
      const _Float16* base = &As[buf][(wm * 32 + mi * 16 + mrow) * A_PAD];
      h8 lo = *(const h8*)(base + hi * 8);
      h8 hh = *(const h8*)(base + 16 + hi * 8);
      af[mi] = __builtin_shufflevector(lo, hh, 0,1,2,3,4,5,6,7,8,9,10,11,12,13,14,15);
    }
#pragma unroll
    for (int ni = 0; ni < 4; ++ni) {
      const _Float16* base = &Bs[buf][wn * 64 + ni * 16 + mrow];
#pragma unroll
      for (int s = 0; s < 16; ++s)
        bfr[ni][s] = base[((s >> 3) * 16 + hi * 8 + (s & 7)) * B_PAD];
    }
#pragma unroll
    for (int mi = 0; mi < 2; ++mi)
#pragma unroll
      for (int ni = 0; ni < 4; ++ni)
        acc[mi][ni] = __builtin_amdgcn_wmma_f32_16x16x32_f16(
            false, af[mi], false, bfr[ni], (short)0, acc[mi][ni], false, false);
    buf ^= 1;
  }

  // ---- epilogue: C layout lane=(N | M+8 half), VGPR r = M row ----
  const int nloc = lane & 15;
  const int moff = (lane >> 4) * 8;
#pragma unroll
  for (int mi = 0; mi < 2; ++mi)
#pragma unroll
    for (int ni = 0; ni < 4; ++ni)
#pragma unroll
      for (int r = 0; r < 8; ++r) {
        size_t row = rowBase + wm * 32 + mi * 16 + moff + r;
        int    col = colBase + wn * 64 + ni * 16 + nloc;
        size_t o   = row * (size_t)N + col;
        float  v   = acc[mi][ni][r];
        if (EPI == 1) { float u = e1[o]; v = gelu_erf(v + Dv[col] * u) + u; }
        else if (EPI == 2) { v += e1[o]; }
        Cout[o] = v;
      }
}

// ---------------------------------------------------------------------------
// LayerNorm over last dim = 1024, one block per token row
// ---------------------------------------------------------------------------
__global__ __launch_bounds__(256) void ln_kernel(
    const float* __restrict__ x, const float* __restrict__ w,
    const float* __restrict__ b, float* __restrict__ out) {
  const int N = DIM;
  size_t row = blockIdx.x;
  const float* xr = x + row * (size_t)N;
  float s = 0.f, s2 = 0.f;
  for (int i = threadIdx.x; i < N; i += 256) { float v = xr[i]; s += v; s2 += v * v; }
  __shared__ float rs[256], rq[256];
  rs[threadIdx.x] = s; rq[threadIdx.x] = s2;
  __syncthreads();
  for (int off = 128; off > 0; off >>= 1) {
    if (threadIdx.x < off) { rs[threadIdx.x] += rs[threadIdx.x + off];
                             rq[threadIdx.x] += rq[threadIdx.x + off]; }
    __syncthreads();
  }
  float mu   = rs[0] * (1.0f / N);
  float var  = rq[0] * (1.0f / N) - mu * mu;
  float rstd = rsqrtf(var + 1e-5f);
  for (int i = threadIdx.x; i < N; i += 256)
    out[row * (size_t)N + i] = (xr[i] - mu) * rstd * w[i] + b[i];
}

// ---------------------------------------------------------------------------
// ZOH discretization:  lam_bar = exp(Lam*dt), gamma = (lam_bar-1)/Lam
// ---------------------------------------------------------------------------
__global__ void prep_kernel(const float* __restrict__ Lre, const float* __restrict__ Lim,
                            const float* __restrict__ logstep,
                            float2* __restrict__ lam, float2* __restrict__ gam) {
  int p = blockIdx.x * 256 + threadIdx.x;
  if (p >= PST) return;
  float dt = expf(logstep[p]);
  float lr = Lre[p] * dt, li = Lim[p] * dt;
  float er = expf(lr);
  float lbr = er * cosf(li), lbi = er * sinf(li);
  lam[p] = make_float2(lbr, lbi);
  float ar = lbr - 1.0f, ai = lbi;
  float cr = Lre[p], ci = Lim[p];
  float inv = 1.0f / (cr * cr + ci * ci);
  gam[p] = make_float2((ar * cr + ai * ci) * inv, (ai * cr - ar * ci) * inv);
}

// WB[K=h=1024][N=1024], interleaved cols: n=2p -> Re(gamma_p*B[p,h]), n=2p+1 -> Im
__global__ void make_wb_kernel(const float* __restrict__ Bre, const float* __restrict__ Bim,
                               const float2* __restrict__ gam, _Float16* __restrict__ dst) {
  int idx = blockIdx.x * 256 + threadIdx.x;
  if (idx >= DIM * DIM) return;
  int n = idx & (DIM - 1);
  int h = idx >> 10;
  int p = n >> 1;
  float2 g = gam[p];
  float br = Bre[p * DIM + h], bi = Bim[p * DIM + h];
  float v = (n & 1) ? (g.x * bi + g.y * br) : (g.x * br - g.y * bi);
  dst[idx] = (_Float16)v;
}

// WC[K=2048][N=1024], K interleave k=4p+q: q0: 2C_re[h,p]  q1: -2C_im[h,p]
//                                          q2: 2C_re[h,512+p]  q3: -2C_im[h,512+p]
__global__ void make_wc_kernel(const float* __restrict__ Cre, const float* __restrict__ Cim,
                               _Float16* __restrict__ dst) {
  int idx = blockIdx.x * 256 + threadIdx.x;
  if (idx >= 2 * DIM * DIM) return;
  int h = idx & (DIM - 1);
  int k = idx >> 10;
  int p = k >> 2;
  int q = k & 3;
  int pp = (q >> 1) ? (PST + p) : p;
  float v = (q & 1) ? -2.0f * Cim[(size_t)h * DIM + pp]
                    :  2.0f * Cre[(size_t)h * DIM + pp];
  dst[idx] = (_Float16)v;
}

// dst[K=Kd][N=Nd] = (f16) src[n][k]  with src (Nd,Kd) row-major
__global__ void conv_transpose_kernel(const float* __restrict__ src,
                                      _Float16* __restrict__ dst, int Nd, int Kd) {
  int idx = blockIdx.x * 256 + threadIdx.x;
  if (idx >= Nd * Kd) return;
  int n = idx % Nd;
  int k = idx / Nd;
  dst[idx] = (_Float16)src[(size_t)n * Kd + k];
}

// Bidirectional diagonal complex scan: one thread per (dir, batch, p)
// Bu[t][2p..2p+1] = (re,im);  xs[t][4p + 2*dir .. +1] = state (re,im)
__global__ __launch_bounds__(256) void scan_kernel(
    const float* __restrict__ Bu, const float2* __restrict__ lam,
    float* __restrict__ xs) {
  int tid = blockIdx.x * 256 + threadIdx.x;        // 0..4095
  int dir = tid >> 11;
  int rem = tid & 2047;
  int b   = rem >> 9;
  int p   = rem & (PST - 1);
  float2 lb = lam[p];
  size_t tokBase = (size_t)b * LSEQ;
  int kOut = 4 * p + 2 * dir;
  float sr = 0.f, si = 0.f;
  for (int step = 0; step < LSEQ; ++step) {
    int l = dir ? (LSEQ - 1 - step) : step;
    size_t t = tokBase + l;
    float2 bu = *(const float2*)(Bu + t * (size_t)DIM + 2 * p);
    float nr = lb.x * sr - lb.y * si + bu.x;
    float ni = lb.x * si + lb.y * sr + bu.y;
    sr = nr; si = ni;
    *(float2*)(xs + t * (size_t)(2 * DIM) + kOut) = make_float2(sr, si);
  }
}

// GEGLU: gg[t,j] = h[t,j] * gelu(h[t,1024+j])
__global__ void geglu_kernel(const float* __restrict__ h, float* __restrict__ gg) {
  size_t idx = (size_t)blockIdx.x * 256 + threadIdx.x;
  size_t t = idx >> 10;
  int    j = (int)(idx & (DIM - 1));
  float a = h[t * (size_t)(2 * DIM) + j];
  float g = h[t * (size_t)(2 * DIM) + DIM + j];
  gg[idx] = a * gelu_erf(g);
}

// ---------------------------------------------------------------------------
extern "C" void kernel_launch(void* const* d_in, const int* in_sizes, int n_in,
                              void* d_out, int out_size, void* d_ws, size_t ws_size,
                              hipStream_t stream) {
  (void)in_sizes; (void)n_in; (void)out_size; (void)ws_size;
  const float* x        = (const float*)d_in[0];
  const float* W_in     = (const float*)d_in[1];
  const float* W_out    = (const float*)d_in[2];
  const float* ln1_w    = (const float*)d_in[3];
  const float* ln1_b    = (const float*)d_in[4];
  const float* ln2_w    = (const float*)d_in[5];
  const float* ln2_b    = (const float*)d_in[6];
  const float* Lam_re   = (const float*)d_in[7];
  const float* Lam_im   = (const float*)d_in[8];
  const float* log_step = (const float*)d_in[9];
  const float* B_re     = (const float*)d_in[10];
  const float* B_im     = (const float*)d_in[11];
  const float* C_re     = (const float*)d_in[12];
  const float* C_im     = (const float*)d_in[13];
  const float* Dvec     = (const float*)d_in[14];
  const float* ff_enc   = (const float*)d_in[15];
  const float* ff_dec   = (const float*)d_in[16];
  float* out = (float*)d_out;

  char* ws = (char*)d_ws;
  size_t off = 0;
  auto take = [&](size_t bytes) -> char* {
    char* p = ws + off; off += (bytes + 255) & ~(size_t)255; return p;
  };
  _Float16* W1t  = (_Float16*)take((size_t)DIN * DIM * 2);
  _Float16* WB   = (_Float16*)take((size_t)DIM * DIM * 2);
  _Float16* WC   = (_Float16*)take((size_t)2 * DIM * DIM * 2);
  _Float16* Wenc = (_Float16*)take((size_t)DIM * 2 * DIM * 2);
  _Float16* Wdec = (_Float16*)take((size_t)DIM * DIM * 2);
  _Float16* Wout = (_Float16*)take((size_t)DIM * DIN * 2);
  float2*   lam  = (float2*)take(PST * sizeof(float2));
  float2*   gam  = (float2*)take(PST * sizeof(float2));
  float* buf1 = (float*)take((size_t)TOKENS * DIM * 4);      // h1 -> gg
  float* buf2 = (float*)take((size_t)TOKENS * DIM * 4);      // fx1 -> fx2
  float* buf3 = (float*)take((size_t)TOKENS * DIM * 4);      // Bu -> x2 -> x3
  float* buf4 = (float*)take((size_t)TOKENS * 2 * DIM * 4);  // xs -> henc

  // weight prep
  prep_kernel<<<2, 256, 0, stream>>>(Lam_re, Lam_im, log_step, lam, gam);
  conv_transpose_kernel<<<(DIM * DIN) / 256, 256, 0, stream>>>(W_in,  W1t,  DIM,     DIN);
  conv_transpose_kernel<<<(2 * DIM * DIM) / 256, 256, 0, stream>>>(ff_enc, Wenc, 2 * DIM, DIM);
  conv_transpose_kernel<<<(DIM * DIM) / 256, 256, 0, stream>>>(ff_dec, Wdec, DIM,     DIM);
  conv_transpose_kernel<<<(DIN * DIM) / 256, 256, 0, stream>>>(W_out,  Wout, DIN,     DIM);
  make_wb_kernel<<<(DIM * DIM) / 256, 256, 0, stream>>>(B_re, B_im, gam, WB);
  make_wc_kernel<<<(2 * DIM * DIM) / 256, 256, 0, stream>>>(C_re, C_im, WC);

  const dim3 gN1024(TOKENS / BM, DIM / BN);
  // h1 = x @ W_in.T
  gemm_f16_kernel<0><<<gN1024, 256, 0, stream>>>(x, W1t, buf1, TOKENS, DIN, DIM, nullptr, nullptr);
  // fx1 = LN1(h1)
  ln_kernel<<<TOKENS, 256, 0, stream>>>(buf1, ln1_w, ln1_b, buf2);
  // Bu = fx1 @ B_bar.T (re/im interleaved)
  gemm_f16_kernel<0><<<gN1024, 256, 0, stream>>>(buf2, WB, buf3, TOKENS, DIM, DIM, nullptr, nullptr);
  // bidirectional scan: Bu -> xs
  scan_kernel<<<16, 256, 0, stream>>>(buf3, lam, buf4);
  // x2 = gelu(2Re(C xs) + D*fx1) + fx1
  gemm_f16_kernel<1><<<gN1024, 256, 0, stream>>>(buf4, WC, buf3, TOKENS, 2 * DIM, DIM, buf2, Dvec);
  // fx2 = LN2(x2)
  ln_kernel<<<TOKENS, 256, 0, stream>>>(buf3, ln2_w, ln2_b, buf2);
  // henc = fx2 @ ff_enc.T
  gemm_f16_kernel<0><<<dim3(TOKENS / BM, (2 * DIM) / BN), 256, 0, stream>>>(
      buf2, Wenc, buf4, TOKENS, DIM, 2 * DIM, nullptr, nullptr);
  // gg = a * gelu(g)
  geglu_kernel<<<((size_t)TOKENS * DIM) / 256, 256, 0, stream>>>(buf4, buf1);
  // x3 = gg @ ff_dec.T + fx2
  gemm_f16_kernel<2><<<gN1024, 256, 0, stream>>>(buf1, Wdec, buf3, TOKENS, DIM, DIM, buf2, nullptr);
  // out = x3 @ W_out.T
  gemm_f16_kernel<0><<<dim3(TOKENS / BM, DIN / BN), 256, 0, stream>>>(
      buf3, Wout, out, TOKENS, DIM, DIN, nullptr, nullptr);
}
